// GIN_635655160273
// MI455X (gfx1250) — matrix operand order repaired and checked by reading, not compiled, not run on hardware
//
#include <hip/hip_runtime.h>
#include <hip/hip_bf16.h>

#define N_NODES 100000
#define N_EDGES 600000
#define DIM     128

typedef __attribute__((ext_vector_type(16))) __bf16 v16bf;
typedef __attribute__((ext_vector_type(8)))  float  v8f;
typedef int async_i4 __attribute__((vector_size(16)));   // int4 for async-LDS builtin
typedef __attribute__((address_space(1))) async_i4 async_i4_g;
typedef __attribute__((address_space(3))) async_i4 async_i4_l;

union Frag { v16bf v; uint4 q[2]; };

__device__ __forceinline__ unsigned short f2bf(float f) {
    unsigned int u = __float_as_uint(f);
    unsigned int r = u + 0x7FFFu + ((u >> 16) & 1u);   // round-to-nearest-even
    return (unsigned short)(r >> 16);
}

// ---------------- utility kernels ----------------

__global__ void k_zero_f4(float4* __restrict__ p, int n4) {
    int i = blockIdx.x * blockDim.x + threadIdx.x;
    if (i < n4) p[i] = float4{0.f, 0.f, 0.f, 0.f};
}

__global__ void k_deg(const int* __restrict__ dst, float* __restrict__ deg, int nE) {
    int e = blockIdx.x * blockDim.x + threadIdx.x;
    if (e < nE) atomicAdd(deg + dst[e], 1.0f);
}

__global__ void k_invdeg(float* __restrict__ d, int n) {
    int i = blockIdx.x * blockDim.x + threadIdx.x;
    if (i < n) d[i] = 1.0f / fmaxf(d[i], 1.0f);
}

// one wave per edge: gather h[src] (float4/lane), atomic scatter-add to sum[dst]
__global__ void k_scatter(const float* __restrict__ h, const int* __restrict__ src,
                          const int* __restrict__ dst, float* __restrict__ sum, int nE) {
    unsigned int tid  = blockIdx.x * blockDim.x + threadIdx.x;
    unsigned int e    = tid >> 5;
    unsigned int lane = tid & 31;
    if (e >= (unsigned int)nE) return;
    int s = src[e];
    int d = dst[e];
    float4 v = *(const float4*)(h + (size_t)s * DIM + lane * 4);
    float* o = sum + (size_t)d * DIM + lane * 4;
    atomicAdd(o + 0, v.x); atomicAdd(o + 1, v.y);
    atomicAdd(o + 2, v.z); atomicAdd(o + 3, v.w);
}

// z = bf16( (1+eps)*h + sum * invdeg ), 8 channels / thread, 16B store
__global__ void k_combine(const float* __restrict__ h, const float* __restrict__ sum,
                          const float* __restrict__ invdeg, const float* __restrict__ epsArr,
                          int layer, unsigned short* __restrict__ z, int nNodes) {
    unsigned int tid  = blockIdx.x * blockDim.x + threadIdx.x;
    unsigned int node = tid >> 4;
    unsigned int c    = (tid & 15) * 8;
    if (node >= (unsigned int)nNodes) return;
    float inv = invdeg[node];
    float ep  = 1.0f + epsArr[layer];
    const float* hp = h   + (size_t)node * DIM + c;
    const float* sp = sum + (size_t)node * DIM + c;
    union { unsigned short s[8]; uint4 q; } o;
#pragma unroll
    for (int j = 0; j < 8; ++j)
        o.s[j] = f2bf(ep * hp[j] + sp[j] * inv);
    *(uint4*)(z + (size_t)node * DIM + c) = o.q;
}

// pack fp32 KxN row-major weight into this file's contiguous-k bf16 B-fragment layout:
// Bp[((ks*NT + t)*32 + lane)*16 + e]  <-  W[ks*32 + (lane>>4)*16 + e][t*16 + (lane&15)]
__global__ void k_pack(const float* __restrict__ W, unsigned short* __restrict__ out,
                       int K, int N) {
    int p = blockIdx.x * blockDim.x + threadIdx.x;
    if (p >= K * N) return;
    int e = p & 15;
    int L = (p >> 4) & 31;
    int rest = p >> 9;
    int nt = N >> 4;
    int t = rest % nt;
    int s = rest / nt;
    int k = s * 32 + (L >> 4) * 16 + e;
    int n = t * 16 + (L & 15);
    out[p] = f2bf(W[(size_t)k * N + n]);
}

// ---------------- WMMA GEMM: out = leaky_relu(A(MxK) @ W(KxN) + bias) ----------------
// A: bf16 row-major (global). B: pre-packed bf16 fragments, staged to LDS per block
// via async global->LDS DMA, then double-buffered through VGPRs so LDS latency is
// hidden behind the WMMA pipeline. One wave computes a 32-row M-tile over all N.

template<int K, int N, bool OUTF32>
__global__ void __launch_bounds__(128) k_gemm(const unsigned short* __restrict__ A,
                                              const unsigned short* __restrict__ Bp,
                                              const float* __restrict__ bias,
                                              void* __restrict__ outp, int M) {
    constexpr int KS   = K / 32;
    constexpr int NT   = N / 16;
    constexpr int ELTS = K * N;                 // bf16 elements in packed weight
    __shared__ __align__(32) unsigned short bsh[ELTS];

    int lane = threadIdx.x & 31;
    int wave = threadIdx.x >> 5;

    // ---- stage packed B into LDS (async global->LDS DMA when available) ----
    constexpr int CHUNKS = ELTS / 8;            // 16-byte chunks
#pragma unroll 1
    for (int c = threadIdx.x; c < CHUNKS; c += 128) {
#if __has_builtin(__builtin_amdgcn_global_load_async_to_lds_b128)
        __builtin_amdgcn_global_load_async_to_lds_b128(
            (async_i4_g*)(Bp + c * 8),
            (async_i4_l*)(bsh + c * 8), 0, 0);
#else
        *(uint4*)(bsh + c * 8) = *(const uint4*)(Bp + c * 8);
#endif
    }
#if __has_builtin(__builtin_amdgcn_global_load_async_to_lds_b128)
    asm volatile("s_wait_asynccnt 0x0" ::: "memory");
#endif
    __syncthreads();

    int tile    = blockIdx.x * 4 + wave;        // each wave: 32 rows
    int rowBase = tile * 32;
    if (rowBase >= M) return;                   // wave-uniform: EXEC stays all-ones
    int mrow = lane & 15;
    int half = lane >> 4;

    const unsigned short* arow0 = A + (size_t)(rowBase + mrow) * K;
    const unsigned short* arow1 = arow0 + (size_t)16 * K;

    v8f acc[2][NT] = {};

    // B fragment double buffer: prefetch j+1 while WMMAs consume j
    Frag bfr[2];
    {
        const unsigned short* bp = bsh + (size_t)lane * 16;
        bfr[0].q[0] = *(const uint4*)(bp);
        bfr[0].q[1] = *(const uint4*)(bp + 8);
    }

#pragma unroll
    for (int ks = 0; ks < KS; ++ks) {
        Frag a0, a1;
        // A 16x32 bf16 layout: lanes 0-15 hold K 0-15, lanes 16-31 hold K 16-31 (matches packed B)
        a0.q[0] = *(const uint4*)(arow0 + ks * 32 + half * 16);
        a0.q[1] = *(const uint4*)(arow0 + ks * 32 + half * 16 + 8);
        a1.q[0] = *(const uint4*)(arow1 + ks * 32 + half * 16);
        a1.q[1] = *(const uint4*)(arow1 + ks * 32 + half * 16 + 8);
#pragma unroll
        for (int t = 0; t < NT; ++t) {
            const int j = ks * NT + t;
            if (j + 1 < KS * NT) {              // prefetch next B fragment (ds_load_b128 x2)
                const unsigned short* bp = bsh + ((size_t)(j + 1) * 32 + lane) * 16;
                bfr[(j + 1) & 1].q[0] = *(const uint4*)(bp);
                bfr[(j + 1) & 1].q[1] = *(const uint4*)(bp + 8);
            }
            const Frag& b = bfr[j & 1];
            acc[0][t] = __builtin_amdgcn_wmma_f32_16x16x32_bf16(
                            false, a0.v, false, b.v, (short)0, acc[0][t], false, false);
            acc[1][t] = __builtin_amdgcn_wmma_f32_16x16x32_bf16(
                            false, a1.v, false, b.v, (short)0, acc[1][t], false, false);
        }
    }

#pragma unroll
    for (int m = 0; m < 2; ++m) {
#pragma unroll
        for (int t = 0; t < NT; ++t) {
            int col = t * 16 + mrow;
            float bs = bias[col];
#pragma unroll
            for (int r = 0; r < 8; ++r) {
                int row = rowBase + m * 16 + half * 8 + r;  // C/D: vgpr r -> M = half*8 + r
                float v = acc[m][t][r] + bs;
                v = v > 0.f ? v : 0.01f * v;                // leaky_relu
                if (OUTF32) ((float*)outp)[(size_t)row * N + col] = v;
                else        ((unsigned short*)outp)[(size_t)row * N + col] = f2bf(v);
            }
        }
    }
}

// ---------------- host-side orchestration ----------------

extern "C" void kernel_launch(void* const* d_in, const int* in_sizes, int n_in,
                              void* d_out, int out_size, void* d_ws, size_t ws_size,
                              hipStream_t stream) {
    (void)in_sizes; (void)n_in; (void)out_size; (void)ws_size;

    const float* x    = (const float*)d_in[0];
    const int*   src  = (const int*)  d_in[1];
    const int*   dst  = (const int*)  d_in[2];
    const float* eps  = (const float*)d_in[3];
    const float* w1s  = (const float*)d_in[4];
    const float* b1s  = (const float*)d_in[5];
    const float* w2s  = (const float*)d_in[6];
    const float* b2s  = (const float*)d_in[7];
    const float* wf1  = (const float*)d_in[8];
    const float* bf1  = (const float*)d_in[9];
    const float* wf2  = (const float*)d_in[10];
    const float* bf2  = (const float*)d_in[11];

    char* ws = (char*)d_ws;
    float*          sum    = (float*)(ws);                          // 51.2 MB
    float*          hbuf   = (float*)(ws + 51200000);               // 51.2 MB
    unsigned short* z      = (unsigned short*)(ws + 102400000);     // 25.6 MB
    unsigned short* act    = (unsigned short*)(ws + 128000000);     // 25.6 MB
    float*          invdeg = (float*)(ws + 153600000);              // 0.4 MB
    unsigned short* wpack  = (unsigned short*)(ws + 154000000);     // 0.22 MB

    const int TB = 256;

    // ---- pack all weights to bf16 B-fragment layout ----
    for (int l = 0; l < 3; ++l) {
        k_pack<<<(16384 + TB - 1) / TB, TB, 0, stream>>>(w1s + (size_t)l * 16384,
                                                         wpack + (size_t)l * 16384, 128, 128);
        k_pack<<<(16384 + TB - 1) / TB, TB, 0, stream>>>(w2s + (size_t)l * 16384,
                                                         wpack + 49152 + (size_t)l * 16384, 128, 128);
    }
    k_pack<<<(8192 + TB - 1) / TB, TB, 0, stream>>>(wf1, wpack + 98304, 128, 64);
    k_pack<<<(4096 + TB - 1) / TB, TB, 0, stream>>>(wf2, wpack + 98304 + 8192, 64, 64);

    // ---- degree -> 1/max(deg,1), computed once ----
    k_zero_f4<<<(N_NODES / 4 + TB - 1) / TB, TB, 0, stream>>>((float4*)invdeg, N_NODES / 4);
    k_deg<<<(N_EDGES + TB - 1) / TB, TB, 0, stream>>>(dst, invdeg, N_EDGES);
    k_invdeg<<<(N_NODES + TB - 1) / TB, TB, 0, stream>>>(invdeg, N_NODES);

    const int sum4       = N_NODES * DIM / 4;                 // 3.2M float4
    const int scatterThr = N_EDGES * 32;
    const int combineThr = N_NODES * 16;
    const int gemmBlocks = (N_NODES / 32 + 3) / 4;            // 3125 wave-tiles of 32 rows

    const float* hcur = x;
    for (int l = 0; l < 4; ++l) {
        k_zero_f4<<<(sum4 + TB - 1) / TB, TB, 0, stream>>>((float4*)sum, sum4);
        k_scatter<<<(scatterThr + TB - 1) / TB, TB, 0, stream>>>(hcur, src, dst, sum, N_EDGES);
        k_combine<<<(combineThr + TB - 1) / TB, TB, 0, stream>>>(hcur, sum, invdeg, eps, l,
                                                                 z, N_NODES);
        if (l < 3) {
            k_gemm<128, 128, false><<<gemmBlocks, 128, 0, stream>>>(
                z, wpack + (size_t)l * 16384, b1s + (size_t)l * 128, act, N_NODES);
            k_gemm<128, 128, true><<<gemmBlocks, 128, 0, stream>>>(
                act, wpack + 49152 + (size_t)l * 16384, b2s + (size_t)l * 128, hbuf, N_NODES);
            hcur = hbuf;
        } else {
            k_gemm<128, 64, false><<<gemmBlocks, 128, 0, stream>>>(
                z, wpack + 98304, bf1, act, N_NODES);
            k_gemm<64, 64, true><<<gemmBlocks, 128, 0, stream>>>(
                act, wpack + 98304 + 8192, bf2, d_out, N_NODES);
        }
    }
}
